// MultiHeadAttention_80307298500716
// MI455X (gfx1250) — matrix operand order repaired and checked by reading, not compiled, not run on hardware
//
#include <hip/hip_runtime.h>
#include <hip/hip_bf16.h>
#include <math.h>

typedef unsigned short u16;
typedef unsigned int   u32;
typedef __bf16 v16bf __attribute__((ext_vector_type(16)));
typedef float  v8f   __attribute__((ext_vector_type(8)));
typedef u32    v4u   __attribute__((ext_vector_type(4)));

union FragBF { u32 u[8]; v4u q[2]; v16bf v; };

__device__ __forceinline__ u16 f2bf(float f) {
  u32 u = __float_as_uint(f);
  u += 0x7FFFu + ((u >> 16) & 1u);   // round-to-nearest-even
  return (u16)(u >> 16);
}

__device__ __forceinline__ void loadFrag(FragBF& f, const u32* p0, const u32* p1) {
  f.q[0] = *(const v4u*)p0;
  f.q[1] = *(const v4u*)p1;
}

// ---------------------------------------------------------------------------
// Pack W (f32 [1024][1024]) into B-fragment-order u32 layout:
//   PW[nt(16)][ks(32)][j(4)][lane(32)][p(8)],  lane = hi*16 + n
//   value = bf16(W[k][c]) | bf16(W[k+1][c])<<16,  k = ks*32 + (p>>2)*16 + hi*8 + (p&3)*2,
//   c = nt*64 + j*16 + n
// ---------------------------------------------------------------------------
__global__ void pack_w(const float* __restrict__ W, u32* __restrict__ PW) {
  int t = blockIdx.x * 256 + threadIdx.x;        // 524288 total
  int p  = t & 7;
  int l  = (t >> 3) & 31;
  int j  = (t >> 8) & 3;
  int ks = (t >> 10) & 31;
  int nt = t >> 15;
  int hi = l >> 4, n = l & 15;
  int c = nt * 64 + j * 16 + n;
  int k = ks * 32 + ((p >> 2) << 4) + (hi << 3) + ((p & 3) << 1);
  u32 lo = f2bf(W[(size_t)k * 1024 + c]);
  u32 h2 = f2bf(W[(size_t)(k + 1) * 1024 + c]);
  PW[t] = lo | (h2 << 16);
}

// ---------------------------------------------------------------------------
// Pack activations into A-fragment-order u32 layout:
//   PA[rb(256)][ks(32)][lane(32)][p(8)],  lane = hi*16 + m
//   k = ks*32 + (p>>2)*16 + hi*8 + (p&3)*2 ; row = rb*16 + m
// ---------------------------------------------------------------------------
__global__ void pack_a_f32(const float* __restrict__ X, u32* __restrict__ PA) {
  int t = blockIdx.x * 256 + threadIdx.x;        // 2097152 total
  int p  = t & 7;
  int l  = (t >> 3) & 31;
  int ks = (t >> 8) & 31;
  int rb = t >> 13;
  int hi = l >> 4, m = l & 15;
  int row = rb * 16 + m;
  int k = ks * 32 + ((p >> 2) << 4) + (hi << 3) + ((p & 3) << 1);
  u32 lo = f2bf(X[(size_t)row * 1024 + k]);
  u32 h2 = f2bf(X[(size_t)row * 1024 + k + 1]);
  PA[t] = lo | (h2 << 16);
}

__global__ void pack_a_bf16(const u16* __restrict__ X, u32* __restrict__ PA) {
  int t = blockIdx.x * 256 + threadIdx.x;
  int p  = t & 7;
  int l  = (t >> 3) & 31;
  int ks = (t >> 8) & 31;
  int rb = t >> 13;
  int hi = l >> 4, m = l & 15;
  int row = rb * 16 + m;
  int k = ks * 32 + ((p >> 2) << 4) + (hi << 3) + ((p & 3) << 1);
  PA[t] = ((const u32*)X)[(size_t)row * 512 + (k >> 1)];  // (k,k+1) adjacent
}

// ---------------------------------------------------------------------------
// GEMM: [4096 x 1024] @ [1024 x 1024], bf16 in / f32 accumulate. LDS-free.
// Block 256 = 8 waves; each wave owns 32 rows (2 row blocks sharing the same
// B fragments) x 64 cols; 32 K-steps x 8 WMMA.
// mode 0: Q proj + RoPE -> outBf [B,H,S,64]
// mode 1: K proj + RoPE -> outBf [B,H,S,64]
// mode 2: V proj -> packed/transposed PVt u32 [bh(32)][dout(64)][kp(1024)]
// mode 3: out proj -> outF f32 [4096][1024]
// ---------------------------------------------------------------------------
__global__ __launch_bounds__(256) void gemm_bf16(
    const u32* __restrict__ PA, const u32* __restrict__ PW,
    u16* __restrict__ outBf, u32* __restrict__ outPV,
    float* __restrict__ outF, int mode)
{
  const int tid  = threadIdx.x;
  const int wave = tid >> 5;
  const int lane = tid & 31;
  const int nQ   = lane & 15;
  const int hi   = lane >> 4;
  const int rb0  = blockIdx.x * 16 + wave * 2;  // two 16-row blocks per wave
  const int nt   = blockIdx.y;                  // 64-col tile == head (modes 0..2)

  v8f acc[2][4];
  #pragma unroll
  for (int i = 0; i < 2; ++i)
    #pragma unroll
    for (int j = 0; j < 4; ++j)
      #pragma unroll
      for (int e = 0; e < 8; ++e) acc[i][j][e] = 0.0f;

  const u32* aBase0 = PA + ((size_t)rb0 * 1024 + lane) * 8;   // + ks*256
  const u32* aBase1 = aBase0 + 8192;                          // next row block
  const u32* bBase  = PW + ((size_t)nt * 4096 + lane) * 8;    // + ks*1024 + j*256

  for (int ks = 0; ks < 32; ++ks) {
    const u32* ap0 = aBase0 + ks * 256;
    const u32* ap1 = aBase1 + ks * 256;
    const u32* bp  = bBase  + ks * 1024;
    if (ks + 1 < 32) {
      __builtin_prefetch((const void*)(ap0 + 256), 0, 1);
      __builtin_prefetch((const void*)(ap1 + 256), 0, 1);
      __builtin_prefetch((const void*)(bp + 1024), 0, 1);
    }
    // load ALL fragments of this K-step into distinct registers first
    FragBF a0, a1, b[4];
    loadFrag(a0, ap0, ap0 + 4);
    loadFrag(a1, ap1, ap1 + 4);
    #pragma unroll
    for (int j = 0; j < 4; ++j) loadFrag(b[j], bp + j * 256, bp + j * 256 + 4);
    // then one burst of 8 WMMAs (B fragments reused by both row blocks)
    #pragma unroll
    for (int j = 0; j < 4; ++j) {
      acc[0][j] = __builtin_amdgcn_wmma_f32_16x16x32_bf16(
          false, a0.v, false, b[j].v, (short)0, acc[0][j], false, false);
      acc[1][j] = __builtin_amdgcn_wmma_f32_16x16x32_bf16(
          false, a1.v, false, b[j].v, (short)0, acc[1][j], false, false);
    }
  }

  // ---- epilogue ----
  #pragma unroll
  for (int rbi = 0; rbi < 2; ++rbi) {
    const int rb = rb0 + rbi;
    if (mode <= 1) {  // RoPE: partner d <-> d+32 lives in acc[.][j] <-> acc[.][j+2]
      #pragma unroll
      for (int j = 0; j < 2; ++j) {
        int dlow = j * 16 + nQ;                               // 0..31
        float invf = exp2f(-(float)dlow * 0.415241011861f);   // 10000^{-d/32}
        #pragma unroll
        for (int r = 0; r < 8; ++r) {
          int t = rb * 16 + r + 8 * hi;
          int b = t >> 11, s = t & 2047;
          float sn, cs;
          __sincosf((float)s * invf, &sn, &cs);
          float x0 = acc[rbi][j][r], x1 = acc[rbi][j + 2][r];
          int base = ((b * 16 + nt) * 2048 + s) * 64;
          outBf[base + dlow]      = f2bf(x0 * cs - x1 * sn);
          outBf[base + dlow + 32] = f2bf(x1 * cs + x0 * sn);
        }
      }
    } else if (mode == 2) {  // pack adjacent key rows (same lane) into PVt
      int t0 = rb * 16 + 8 * hi;
      #pragma unroll
      for (int j = 0; j < 4; ++j) {
        int dout = j * 16 + nQ;
        #pragma unroll
        for (int i = 0; i < 4; ++i) {
          int t = t0 + 2 * i;
          int b = t >> 11, s = t & 2047;
          u32 val = (u32)f2bf(acc[rbi][j][2 * i]) |
                    ((u32)f2bf(acc[rbi][j][2 * i + 1]) << 16);
          outPV[((size_t)(b * 16 + nt) * 64 + dout) * 1024 + (s >> 1)] = val;
        }
      }
    } else {
      #pragma unroll
      for (int j = 0; j < 4; ++j)
        #pragma unroll
        for (int r = 0; r < 8; ++r) {
          int t = rb * 16 + r + 8 * hi;
          outF[(size_t)t * 1024 + nt * 64 + j * 16 + nQ] = acc[rbi][j][r];
        }
    }
  }
}

// ---------------------------------------------------------------------------
// Flash attention (causal, online softmax). Block = (b*H+h, 64-q tile),
// 128 threads = 4 waves, wave owns 16 q rows. Key step 32.
// Q,K: bf16 [B,H,S,64] natural (fragment pairs are along d => contiguous).
// V:   packed PVt u32 [bh][dout][key-pair].
// Only P goes through LDS (C-layout -> A-layout), double-buffered.
// ---------------------------------------------------------------------------
__global__ __launch_bounds__(128) void flash_attn(
    const u32* __restrict__ Qg, const u32* __restrict__ Kg,
    const u32* __restrict__ PVt, u16* __restrict__ ctx)
{
  __shared__ u32 PsU[4 * 2 * 272];   // per-wave double-buffered P tile (stride 17 u32)

  const int tid  = threadIdx.x;
  const int wave = tid >> 5;
  const int lane = tid & 31;
  const int nQ   = lane & 15;
  const int hi   = lane >> 4;
  const int bh   = blockIdx.x;       // b*16 + h
  const int qt   = blockIdx.y;
  const int qm0  = qt * 64;

  const u32* Qh = Qg  + (size_t)bh * 65536;   // [2048][32] u32
  const u32* Kh = Kg  + (size_t)bh * 65536;
  const u32* Vh = PVt + (size_t)bh * 65536;   // [64][1024] u32

  // Q A-fragments for this wave, straight from global (16B-aligned runs)
  FragBF qf[2];
  {
    const int qrow = qm0 + wave * 16 + nQ;
    #pragma unroll
    for (int d0h = 0; d0h < 2; ++d0h) {
      const u32* base = Qh + (size_t)qrow * 32 + d0h * 16 + hi * 4;
      loadFrag(qf[d0h], base, base + 8);
    }
  }

  float mstate[8], lstate[8];
  v8f accO[4];
  #pragma unroll
  for (int r = 0; r < 8; ++r) { mstate[r] = -1e30f; lstate[r] = 0.0f; }
  #pragma unroll
  for (int j = 0; j < 4; ++j)
    #pragma unroll
    for (int e = 0; e < 8; ++e) accO[j][e] = 0.0f;

  const int nkt = qt * 2 + 2;   // key tiles covering keys <= qm0+63
  for (int kt = 0; kt < nkt; ++kt) {
    const int k0 = kt * 32;

    // ---- scores = Q @ K^T : load 4 K B-fragments, then 4 WMMAs ----
    FragBF kb[2][2];   // [j][d0h]
    #pragma unroll
    for (int j = 0; j < 2; ++j)
      #pragma unroll
      for (int d0h = 0; d0h < 2; ++d0h) {
        const u32* bp = Kh + (size_t)(k0 + j * 16 + nQ) * 32 + d0h * 16 + hi * 4;
        loadFrag(kb[j][d0h], bp, bp + 8);
      }
    v8f s[2];
    #pragma unroll
    for (int j = 0; j < 2; ++j) {
      #pragma unroll
      for (int e = 0; e < 8; ++e) s[j][e] = 0.0f;
      s[j] = __builtin_amdgcn_wmma_f32_16x16x32_bf16(
          false, qf[0].v, false, kb[j][0].v, (short)0, s[j], false, false);
      s[j] = __builtin_amdgcn_wmma_f32_16x16x32_bf16(
          false, qf[1].v, false, kb[j][1].v, (short)0, s[j], false, false);
    }

    // ---- online softmax (1/sqrt(64) folded); P -> per-wave LDS tile ----
    u32* PsW = &PsU[wave * 544 + (kt & 1) * 272];
    u16* PsS = (u16*)PsW;
    #pragma unroll
    for (int r = 0; r < 8; ++r) {
      int qg = qm0 + wave * 16 + r + 8 * hi;
      float sc0 = s[0][r] * 0.125f;
      float sc1 = s[1][r] * 0.125f;
      if (k0 + nQ      > qg) sc0 = -1e30f;
      if (k0 + 16 + nQ > qg) sc1 = -1e30f;
      float mx = fmaxf(sc0, sc1);
      #pragma unroll
      for (int off = 8; off; off >>= 1) mx = fmaxf(mx, __shfl_xor(mx, off, 16));
      float mnew  = fmaxf(mstate[r], mx);
      float alpha = __expf(mstate[r] - mnew);
      float p0 = __expf(sc0 - mnew);
      float p1 = __expf(sc1 - mnew);
      float rs = p0 + p1;
      #pragma unroll
      for (int off = 8; off; off >>= 1) rs += __shfl_xor(rs, off, 16);
      lstate[r] = lstate[r] * alpha + rs;
      mstate[r] = mnew;
      #pragma unroll
      for (int j = 0; j < 4; ++j) accO[j][r] *= alpha;
      int row = r + 8 * hi;
      PsS[row * 34 + nQ]      = f2bf(p0);
      PsS[row * 34 + 16 + nQ] = f2bf(p1);
    }
    __syncthreads();   // ordering fence for the u16-write / u32-read round trip

    // ---- accO += P @ V : load P A-frag + 4 V B-frags, then 4 WMMAs ----
    FragBF pf;
    #pragma unroll
    for (int p = 0; p < 8; ++p) {
      int pairidx = (hi << 3) + p;
      pf.u[p] = PsW[nQ * 17 + pairidx];
    }
    FragBF vb[4];
    #pragma unroll
    for (int j = 0; j < 4; ++j) {
      const u32* bp = Vh + (size_t)(j * 16 + nQ) * 1024 + (k0 >> 1) + hi * 8;
      loadFrag(vb[j], bp, bp + 4);
    }
    #pragma unroll
    for (int j = 0; j < 4; ++j)
      accO[j] = __builtin_amdgcn_wmma_f32_16x16x32_bf16(
          false, pf.v, false, vb[j].v, (short)0, accO[j], false, false);
  }

  // epilogue: normalize, store ctx[B,S,1024] (feature = h*64 + d)
  const int b = bh >> 4, h = bh & 15;
  #pragma unroll
  for (int r = 0; r < 8; ++r) {
    float inv = 1.0f / lstate[r];
    int qg = qm0 + wave * 16 + r + 8 * hi;
    size_t base = (size_t)(b * 2048 + qg) * 1024 + h * 64;
    #pragma unroll
    for (int j = 0; j < 4; ++j)
      ctx[base + j * 16 + nQ] = f2bf(accO[j][r] * inv);
  }
}

// ---------------------------------------------------------------------------
extern "C" void kernel_launch(void* const* d_in, const int* in_sizes, int n_in,
                              void* d_out, int out_size, void* d_ws, size_t ws_size,
                              hipStream_t stream) {
  const float* x  = (const float*)d_in[0];
  const float* Wq = (const float*)d_in[1];
  const float* Wk = (const float*)d_in[2];
  const float* Wv = (const float*)d_in[3];
  const float* Wo = (const float*)d_in[4];

  const size_t MB = 1u << 20;
  char* w = (char*)d_ws;
  u32* PAx   = (u32*)(w + 0 * MB);    // 8 MB : packed A of x
  u32* PWq   = (u32*)(w + 8 * MB);    // 2 MB each
  u32* PWk   = (u32*)(w + 10 * MB);
  u32* PWv   = (u32*)(w + 12 * MB);
  u32* PWo   = (u32*)(w + 14 * MB);
  u16* qb    = (u16*)(w + 16 * MB);   // 8 MB : Q bf16 [B,H,S,64]
  u16* kbuf  = (u16*)(w + 24 * MB);   // 8 MB : K bf16 [B,H,S,64]
  u32* PVt   = (u32*)(w + 32 * MB);   // 8 MB : V packed [bh][dout][kp]
  u16* ctxb  = (u16*)(w + 40 * MB);   // 8 MB : ctx bf16 [B,S,1024]
  u32* PActx = (u32*)(w + 48 * MB);   // 8 MB : packed A of ctx

  pack_a_f32<<<8192, 256, 0, stream>>>(x, PAx);
  pack_w<<<2048, 256, 0, stream>>>(Wq, PWq);
  pack_w<<<2048, 256, 0, stream>>>(Wk, PWk);
  pack_w<<<2048, 256, 0, stream>>>(Wv, PWv);
  pack_w<<<2048, 256, 0, stream>>>(Wo, PWo);

  dim3 gG(16, 16);  // 4096/256 M-tiles x 1024/64 N-tiles
  gemm_bf16<<<gG, 256, 0, stream>>>(PAx, PWq, qb,   nullptr, nullptr, 0);
  gemm_bf16<<<gG, 256, 0, stream>>>(PAx, PWk, kbuf, nullptr, nullptr, 1);
  gemm_bf16<<<gG, 256, 0, stream>>>(PAx, PWv, nullptr, PVt,  nullptr, 2);

  dim3 gA(32, 32);
  flash_attn<<<gA, 128, 0, stream>>>((const u32*)qb, (const u32*)kbuf, PVt, ctxb);

  pack_a_bf16<<<8192, 256, 0, stream>>>(ctxb, PActx);
  gemm_bf16<<<gG, 256, 0, stream>>>(PActx, PWo, nullptr, nullptr, (float*)d_out, 3);
}
